// TimeAttention_13185549598743
// MI455X (gfx1250) — matrix operand order repaired and checked by reading, not compiled, not run on hardware
//
#include <hip/hip_runtime.h>
#include <hip/hip_bf16.h>
#include <math.h>

typedef __attribute__((ext_vector_type(16))) __bf16 v16bf;
typedef __attribute__((ext_vector_type(8)))  float  v8f;
typedef __attribute__((ext_vector_type(8)))  __bf16 bf16x8;
typedef __attribute__((ext_vector_type(2)))  __bf16 bf16x2;

#define TA_B   2
#define TA_NV  32
#define TA_NT  48
#define TA_H   16
#define TA_E   64
#define TA_L   (TA_NV * TA_NT)   // 1536
#define CHUNK  128
#define HALF   64                // softmax granularity (keys)
#define KPAD   72                // kbuf row stride in bf16 elems (144B, 16B-aligned)
#define VPAD   136               // vT   row stride (272B)
#define PPAD   72                // pbuf row stride (144B)

// Load 16 consecutive bf16 (two ds_load_b128) into a WMMA B-operand half-run.
__device__ __forceinline__ v16bf ld_run16(const __bf16* p) {
  bf16x8 a = *(const bf16x8*)(p);
  bf16x8 b = *(const bf16x8*)(p + 8);
  v16bf r;
#pragma unroll
  for (int i = 0; i < 8; ++i) { r[i] = a[i]; r[i + 8] = b[i]; }
  return r;
}

__global__ __launch_bounds__(256)
void ta_attn_kernel(const float* __restrict__ Q, const float* __restrict__ K,
                    const float* __restrict__ V, const float* __restrict__ BE,
                    float* __restrict__ O)
{
  // 54.3 KB static LDS
  __shared__ __align__(16) __bf16 kbuf[CHUNK][KPAD];     // RoPE'd keys, bf16
  __shared__ __align__(16) __bf16 vT[TA_E][VPAD];        // values, transposed
  __shared__ __align__(16) __bf16 pbuf[8][16][PPAD];     // per-wave P staging (16x64)

  const int tid  = threadIdx.x;
  const int wave = tid >> 5;
  const int lane = tid & 31;
  const int lm   = lane & 15;   // M (A) / N (B,C) index
  const int lh   = lane >> 4;   // half: selects K-run / row group

  const int b     = blockIdx.z;
  const int hd    = blockIdx.y;
  const int qbase = blockIdx.x * 128;

  const float sb_diff = 0.125f * BE[hd];          // scale pre-folded into bias
  const float sb_same = 0.125f * BE[TA_H + hd];
  const float scale = 0.125f;                     // 1/sqrt(64)
  const float LN1E4_D16 = 0.5756462732485114f;    // ln(10000)/16
  const float NEG_BIG = -1e30f;

  // ---- Q tile -> A operands (RoPE on e<32), lane lm owns row qbase+wave*16+lm ----
  const int qrow = qbase + wave * 16 + lm;
  const float* qp = Q + ((size_t)(b * TA_L + qrow) * TA_H + hd) * TA_E;
  const float posq = (float)qrow;
  v16bf qa0, qa1;
#pragma unroll
  for (int run = 0; run < 2; ++run) {
    const int e0 = lh * 16 + run * 8;             // A-run bases: lh*16 and lh*16+8
#pragma unroll
    for (int p2 = 0; p2 < 4; ++p2) {
      const int e = e0 + 2 * p2;                  // e < 32 -> RoPE pair
      const float x1 = qp[e], x2 = qp[e + 1];
      const float th = __expf(-(float)(e >> 1) * LN1E4_D16);
      float sn, cs; __sincosf(posq * th, &sn, &cs);
      qa0[run * 8 + 2 * p2]     = (__bf16)(cs * x1 - sn * x2);
      qa0[run * 8 + 2 * p2 + 1] = (__bf16)(cs * x2 + sn * x1);
    }
#pragma unroll
    for (int j = 0; j < 8; ++j)                   // e in [32,64): no RoPE
      qa1[run * 8 + j] = (__bf16)qp[32 + e0 + j];
  }

  // All-ones B operand for WMMA row-sum of P
  v16bf ones;
#pragma unroll
  for (int i = 0; i < 16; ++i) ones[i] = (__bf16)1.0f;

  // Per-lane row constants: C-layout element r <-> local row r + 8*lh
  int tl[8], vl[8];
#pragma unroll
  for (int r = 0; r < 8; ++r) {
    const int lg = qbase + wave * 16 + 8 * lh + r;
    vl[r] = lg / TA_NT;
    tl[r] = lg % TA_NT;
  }

  v8f o0 = {}, o1 = {}, o2 = {}, o3 = {};         // 16x64 fp32 output accum
  float mrun[8], lrun[8];
#pragma unroll
  for (int r = 0; r < 8; ++r) { mrun[r] = NEG_BIG; lrun[r] = 0.f; }

#pragma unroll 1
  for (int cb = 0; cb < TA_L; cb += CHUNK) {
    __syncthreads();
    // ---- cooperative K chunk load: fp32 -> RoPE -> bf16 LDS (coalesced float2) ----
#pragma unroll
    for (int it = 0; it < 16; ++it) {
      const int pidx = tid + it * 256;            // 128 rows * 32 pairs
      const int row  = pidx >> 5;
      const int e    = (pidx & 31) * 2;
      const int gk   = cb + row;
      const float* kp = K + ((size_t)(b * TA_L + gk) * TA_H + hd) * TA_E + e;
      float x1 = kp[0], x2 = kp[1];
      if (e < 32) {
        const float th = __expf(-(float)(e >> 1) * LN1E4_D16);
        float sn, cs; __sincosf((float)gk * th, &sn, &cs);
        const float r1 = cs * x1 - sn * x2;
        const float r2 = cs * x2 + sn * x1;
        x1 = r1; x2 = r2;
      }
      bf16x2 w; w[0] = (__bf16)x1; w[1] = (__bf16)x2;
      *(bf16x2*)&kbuf[row][e] = w;
    }
    // ---- cooperative V chunk load, transposed into vT[e][key] ----
#pragma unroll
    for (int it = 0; it < 16; ++it) {
      const int pidx = tid + it * 256;
      const int row  = pidx >> 5;
      const int e    = (pidx & 31) * 2;
      const float* vp = V + ((size_t)(b * TA_L + cb + row) * TA_H + hd) * TA_E + e;
      vT[e][row]     = (__bf16)vp[0];
      vT[e + 1][row] = (__bf16)vp[1];
    }
    // ---- prefetch next chunk into cache hierarchy (global_prefetch_b8) ----
    if (cb + CHUNK < TA_L) {
      const int prow = tid >> 1;
      const int poff = (tid & 1) * 32;            // two 128B lines per 256B row
      const size_t pbase = ((size_t)(b * TA_L + cb + CHUNK + prow) * TA_H + hd) * TA_E + poff;
      __builtin_prefetch(K + pbase, 0, 3);
      __builtin_prefetch(V + pbase, 0, 3);
    }
    __syncthreads();

    // ---- compute: two 64-key halves per chunk ----
#pragma unroll
    for (int half = 0; half < 2; ++half) {
      const int base = half * HALF;

      // ---- QK^T scores for four 16-col sub-tiles ----
      v8f c[4];
#pragma unroll
      for (int t = 0; t < 4; ++t) {
        const int krow = base + t * 16 + lm;
        const v16bf kb0 = ld_run16(&kbuf[krow][lh * 16]);
        const v16bf kb1 = ld_run16(&kbuf[krow][32 + lh * 16]);
        v8f acc = {};
        acc = __builtin_amdgcn_wmma_f32_16x16x32_bf16(false, qa0, false, kb0, (short)0, acc, false, false);
        acc = __builtin_amdgcn_wmma_f32_16x16x32_bf16(false, qa1, false, kb1, (short)0, acc, false, false);
        c[t] = acc;
      }

      // ---- bias + mask + scale (in place), incremental tok/var column ids ----
      {
        const int s0 = cb + base + lm;
        int vs = s0 / TA_NT;
        int ts = s0 % TA_NT;
#pragma unroll
        for (int t = 0; t < 4; ++t) {
#pragma unroll
          for (int r = 0; r < 8; ++r) {
            const float sb = (vs == vl[r]) ? sb_same : sb_diff;
            const float x  = fmaf(scale, c[t][r], sb);
            c[t][r] = (ts > tl[r]) ? -__builtin_inff() : x;
          }
          ts += 16;
          if (ts >= TA_NT) { ts -= TA_NT; ++vs; }
        }
      }

      // ---- row max over 64 cols, rescale running state (branch-free) ----
#pragma unroll
      for (int r = 0; r < 8; ++r) {
        float mx = fmaxf(fmaxf(c[0][r], c[1][r]), fmaxf(c[2][r], c[3][r]));
#pragma unroll
        for (int d = 1; d < 16; d <<= 1)
          mx = fmaxf(mx, __shfl_xor(mx, d, 32));
        const float mnew = fmaxf(fmaxf(mrun[r], mx), NEG_BIG);  // clamp: no NaN guards
        const float corr = __expf(mrun[r] - mnew);
        mrun[r] = mnew;
        lrun[r] *= corr;
        o0[r] *= corr; o1[r] *= corr; o2[r] *= corr; o3[r] *= corr;
      }

      // ---- P = exp(x - m), stage to LDS in bf16 ----
#pragma unroll
      for (int t = 0; t < 4; ++t) {
#pragma unroll
        for (int r = 0; r < 8; ++r) {
          const float p = __expf(c[t][r] - mrun[r]);   // exp(-inf)=0 handles mask
          pbuf[wave][8 * lh + r][t * 16 + lm] = (__bf16)p;
        }
      }
      // same-wave DS ordering; block compiler motion + wait the DS counter
      asm volatile("s_wait_dscnt 0" ::: "memory");

      // ---- P*V (two 32-key groups) + WMMA row-sum (P x ones) ----
      v8f ls = {};
#pragma unroll
      for (int g2 = 0; g2 < 2; ++g2) {
        const int koff = g2 * 32;
        v16bf pa;
        {
          bf16x8 a  = *(const bf16x8*)&pbuf[wave][lm][koff + lh * 16];
          bf16x8 bh = *(const bf16x8*)&pbuf[wave][lm][koff + lh * 16 + 8];
#pragma unroll
          for (int i = 0; i < 8; ++i) { pa[i] = a[i]; pa[i + 8] = bh[i]; }
        }
        const int kb = base + koff + lh * 16;
        const v16bf vb0 = ld_run16(&vT[ 0 + lm][kb]);
        const v16bf vb1 = ld_run16(&vT[16 + lm][kb]);
        const v16bf vb2 = ld_run16(&vT[32 + lm][kb]);
        const v16bf vb3 = ld_run16(&vT[48 + lm][kb]);
        o0 = __builtin_amdgcn_wmma_f32_16x16x32_bf16(false, pa, false, vb0,  (short)0, o0, false, false);
        o1 = __builtin_amdgcn_wmma_f32_16x16x32_bf16(false, pa, false, vb1,  (short)0, o1, false, false);
        o2 = __builtin_amdgcn_wmma_f32_16x16x32_bf16(false, pa, false, vb2,  (short)0, o2, false, false);
        o3 = __builtin_amdgcn_wmma_f32_16x16x32_bf16(false, pa, false, vb3,  (short)0, o3, false, false);
        ls = __builtin_amdgcn_wmma_f32_16x16x32_bf16(false, pa, false, ones, (short)0, ls, false, false);
      }
#pragma unroll
      for (int r = 0; r < 8; ++r)
        lrun[r] += ls[r];
    }
  }

  // ---- epilogue: normalize and store (B, L, H, E) ----
#pragma unroll
  for (int r = 0; r < 8; ++r) {
    const float inv = 1.0f / lrun[r];
    const int lg = qbase + wave * 16 + 8 * lh + r;
    float* op = O + ((size_t)(b * TA_L + lg) * TA_H + hd) * TA_E;
    op[ 0 + lm] = o0[r] * inv;
    op[16 + lm] = o1[r] * inv;
    op[32 + lm] = o2[r] * inv;
    op[48 + lm] = o3[r] * inv;
  }
}

extern "C" void kernel_launch(void* const* d_in, const int* in_sizes, int n_in,
                              void* d_out, int out_size, void* d_ws, size_t ws_size,
                              hipStream_t stream) {
  (void)in_sizes; (void)n_in; (void)out_size; (void)d_ws; (void)ws_size;
  const float* Q  = (const float*)d_in[0];
  const float* K  = (const float*)d_in[1];
  const float* V  = (const float*)d_in[2];
  const float* BE = (const float*)d_in[3];
  // d_in[4] = n_vars, d_in[5] = n_tokens (compile-time constants here)
  dim3 grid(TA_L / 128, TA_H, TA_B);   // 12 x 16 x 2
  ta_attn_kernel<<<grid, dim3(256), 0, stream>>>(Q, K, V, BE, (float*)d_out);
}
